// SelfAttV2_1348619731354
// MI455X (gfx1250) — matrix operand order repaired and checked by reading, not compiled, not run on hardware
//
#include <hip/hip_runtime.h>
#include <hip/hip_bf16.h>

typedef __attribute__((ext_vector_type(16))) _Float16 v16h;
typedef __attribute__((ext_vector_type(8)))  float    v8f;

union Frag16 { v16h v; _Float16 h[16]; unsigned u[8]; };
union Frag32 { v8f  v; float f[8]; };

#define DIM   256
#define BATCH 4
#define SEQ   4096
#define NTOK  (BATCH * SEQ)          // 16384
#define QK_SCALE 0.0625f             // 1/sqrt(256)

__device__ __forceinline__ float half_reduce_max(float v) {
    v = fmaxf(v, __shfl_xor(v, 1, 32));
    v = fmaxf(v, __shfl_xor(v, 2, 32));
    v = fmaxf(v, __shfl_xor(v, 4, 32));
    v = fmaxf(v, __shfl_xor(v, 8, 32));
    return v;
}
__device__ __forceinline__ float half_reduce_sum(float v) {
    v += __shfl_xor(v, 1, 32);
    v += __shfl_xor(v, 2, 32);
    v += __shfl_xor(v, 4, 32);
    v += __shfl_xor(v, 8, 32);
    return v;
}

// ---------------------------------------------------------------------------
// Kernel 1: QKV = X @ W^T + b.  One wave -> one 16x16 tile of [16384 x 768].
// Writes Qh (scaled), Kh row-major f16 and Vt transposed [batch][d][s] f16.
// ---------------------------------------------------------------------------
__global__ __launch_bounds__(128) void qkv_proj_kernel(
    const float* __restrict__ X, const float* __restrict__ W,
    const float* __restrict__ bias,
    _Float16* __restrict__ Qh, _Float16* __restrict__ Kh,
    _Float16* __restrict__ Vt)
{
    const int waveId = blockIdx.x * 4 + (threadIdx.x >> 5);
    const int mt = waveId / 48;           // token tile   [0,1024)
    const int nt = waveId % 48;           // feature tile [0,48)
    const int lane = threadIdx.x & 31;
    const int row  = lane & 15;
    const int half = lane >> 4;           // 0: lanes 0-15, 1: lanes 16-31
    const int n    = lane & 15;
    const int e    = nt * 16 + n;         // output feature [0,768)
    const int tokA = mt * 16 + row;

    Frag32 acc;
    #pragma unroll
    for (int j = 0; j < 8; ++j) acc.f[j] = 0.0f;

    #pragma unroll 2
    for (int kb = 0; kb < DIM; kb += 32) {
        Frag16 fa, fb;
        #pragma unroll
        for (int j = 0; j < 8; ++j) {
            // A (16x32 f16): lane=row, dword j holds K pair (ka, ka+1)
            const int ka = (j < 4 ? 0 : 16) + half * 8 + (j & 3) * 2;
            float2 xa = *reinterpret_cast<const float2*>(X + tokA * DIM + kb + ka);
            fa.h[2 * j]     = (_Float16)xa.x;
            fa.h[2 * j + 1] = (_Float16)xa.y;
            // B (32x16 f16) = W^T tile: element (k,n) = W[e][k], k contiguous
            const int kbidx = (j < 4 ? 0 : 16) + half * 8 + (j & 3) * 2;
            float2 wb = *reinterpret_cast<const float2*>(W + e * DIM + kb + kbidx);
            fb.h[2 * j]     = (_Float16)wb.x;
            fb.h[2 * j + 1] = (_Float16)wb.y;
        }
        acc.v = __builtin_amdgcn_wmma_f32_16x16x32_f16(
            false, fa.v, false, fb.v, (short)0, acc.v, false, false);
    }

    const float bv = bias[e];
    // C layout: vgpr j -> row (j + 8*half), col n.
    #pragma unroll
    for (int j = 0; j < 8; ++j) {
        const int tok = mt * 16 + j + half * 8;
        const float val = acc.f[j] + bv;
        if (e < DIM) {
            Qh[tok * DIM + e] = (_Float16)(val * QK_SCALE);
        } else if (e < 2 * DIM) {
            Kh[tok * DIM + (e - DIM)] = (_Float16)val;
        } else {
            const int b = tok >> 12;              // /SEQ
            const int s = tok & (SEQ - 1);
            Vt[b * (DIM * SEQ) + (e - 2 * DIM) * SEQ + s] = (_Float16)val;
        }
    }
}

// ---------------------------------------------------------------------------
// Kernel 2: flash attention.  One wave -> TWO 16-query tiles (32 queries);
// K/V fragments are loaded once and feed a WMMA pair (B-operand reuse).
// Per 32-key chunk: 32 score WMMA + 16 PV WMMA = 48 WMMA on 48 B128 loads.
// ---------------------------------------------------------------------------
__global__ __launch_bounds__(128) void flash_attn_kernel(
    const _Float16* __restrict__ Qh, const _Float16* __restrict__ Kh,
    const _Float16* __restrict__ Vt, float* __restrict__ Out)
{
    __shared__ _Float16 Plds[4][2][16 * 32];   // per-wave, per-qtile P staging
    const int w = threadIdx.x >> 5;
    const int waveId = blockIdx.x * 4 + w;     // [0,512)
    const int qt0 = waveId * 2;                // query tiles qt0, qt0+1
    const int batch = qt0 >> 8;                // 256 tiles per batch
    const int qtok = qt0 * 16;                 // first of 32 query tokens
    const int lane = threadIdx.x & 31;
    const int row  = lane & 15;
    const int half = lane >> 4;
    const int n    = lane & 15;

    // Resident Q fragments: 2 tiles x 8 x (16x32 f16) covering K = 256
    Frag16 qa[2][8];
    #pragma unroll
    for (int q = 0; q < 2; ++q)
        #pragma unroll
        for (int f = 0; f < 8; ++f)
            #pragma unroll
            for (int j = 0; j < 8; ++j) {
                const int ka = (j < 4 ? 0 : 16) + half * 8 + (j & 3) * 2;
                qa[q][f].u[j] = *reinterpret_cast<const unsigned*>(
                    Qh + (qtok + q * 16 + row) * DIM + f * 32 + ka);
            }

    float mrow[2][8], lrow[2][8];
    Frag32 oacc[2][8];
    #pragma unroll
    for (int q = 0; q < 2; ++q)
        #pragma unroll
        for (int j = 0; j < 8; ++j) {
            mrow[q][j] = -1e30f; lrow[q][j] = 0.0f;
            #pragma unroll
            for (int c = 0; c < 8; ++c) oacc[q][c].f[j] = 0.0f;
        }

    const int kTokBase = batch * SEQ;
    const _Float16* __restrict__ Vbase = Vt + batch * (DIM * SEQ);

    for (int kc = 0; kc < SEQ; kc += 32) {
        // ---- scores: sc[keytile][qtile], B fragment shared by both q tiles
        Frag32 sc[2][2];
        #pragma unroll
        for (int t = 0; t < 2; ++t)
            #pragma unroll
            for (int q = 0; q < 2; ++q)
                #pragma unroll
                for (int j = 0; j < 8; ++j) sc[t][q].f[j] = 0.0f;

        #pragma unroll
        for (int t = 0; t < 2; ++t) {
            const int keyTok = kTokBase + kc + t * 16 + n;
            #pragma unroll
            for (int f = 0; f < 8; ++f) {
                Frag16 kb;
                const int ko = f * 32;
                #pragma unroll
                for (int j = 0; j < 8; ++j)
                    kb.u[j] = *reinterpret_cast<const unsigned*>(
                        Kh + keyTok * DIM + ko + (j < 4 ? 0 : 16) + half * 8 + (j & 3) * 2);
                sc[t][0].v = __builtin_amdgcn_wmma_f32_16x16x32_f16(
                    false, qa[0][f].v, false, kb.v, (short)0, sc[t][0].v, false, false);
                sc[t][1].v = __builtin_amdgcn_wmma_f32_16x16x32_f16(
                    false, qa[1][f].v, false, kb.v, (short)0, sc[t][1].v, false, false);
            }
        }

        // ---- online softmax per q tile (rows live in half-wave groups) ----
        #pragma unroll
        for (int q = 0; q < 2; ++q)
            #pragma unroll
            for (int j = 0; j < 8; ++j) {
                const float cm = half_reduce_max(fmaxf(sc[0][q].f[j], sc[1][q].f[j]));
                const float mnew = fmaxf(mrow[q][j], cm);
                const float scale = __expf(mrow[q][j] - mnew);
                mrow[q][j] = mnew;
                const float p0 = __expf(sc[0][q].f[j] - mnew);
                const float p1 = __expf(sc[1][q].f[j] - mnew);
                lrow[q][j] = lrow[q][j] * scale + half_reduce_sum(p0 + p1);
                #pragma unroll
                for (int c = 0; c < 8; ++c) oacc[q][c].f[j] *= scale;
                const int r = j + half * 8;        // C-layout row
                Plds[w][q][r * 32 + n]      = (_Float16)p0;
                Plds[w][q][r * 32 + 16 + n] = (_Float16)p1;
            }
        asm volatile("s_wait_dscnt 0" ::: "memory");   // LDS stores visible

        // ---- reload P as 16x32 A fragments ----
        Frag16 pa[2];
        #pragma unroll
        for (int q = 0; q < 2; ++q)
            #pragma unroll
            for (int j = 0; j < 8; ++j) {
                const int ka = (j < 4 ? 0 : 16) + half * 8 + (j & 3) * 2;
                pa[q].u[j] = *reinterpret_cast<const unsigned*>(
                    &Plds[w][q][row * 32 + ka]);
            }

        // ---- O += P * V, V fragment shared by both q tiles ----
        #pragma unroll
        for (int c = 0; c < 8; ++c) {
            Frag16 vb;
            const int d = c * 16 + n;
            #pragma unroll
            for (int j = 0; j < 8; ++j) {
                const int k = (j < 4 ? 0 : 16) + half * 8 + (j & 3) * 2;   // key pair within chunk
                vb.u[j] = *reinterpret_cast<const unsigned*>(
                    Vbase + d * SEQ + kc + k);
            }
            oacc[0][c].v = __builtin_amdgcn_wmma_f32_16x16x32_f16(
                false, pa[0].v, false, vb.v, (short)0, oacc[0][c].v, false, false);
            oacc[1][c].v = __builtin_amdgcn_wmma_f32_16x16x32_f16(
                false, pa[1].v, false, vb.v, (short)0, oacc[1][c].v, false, false);
        }
    }

    // ---- normalize and store f32 output ----
    #pragma unroll
    for (int q = 0; q < 2; ++q)
        #pragma unroll
        for (int j = 0; j < 8; ++j) {
            const float inv = 1.0f / lrow[q][j];
            const int tok = qtok + q * 16 + j + half * 8;
            #pragma unroll
            for (int c = 0; c < 8; ++c)
                Out[tok * DIM + c * 16 + n] = oacc[q][c].f[j] * inv;
        }
}

extern "C" void kernel_launch(void* const* d_in, const int* in_sizes, int n_in,
                              void* d_out, int out_size, void* d_ws, size_t ws_size,
                              hipStream_t stream) {
    const float* X    = (const float*)d_in[0];   // [4,4096,256]
    const float* W    = (const float*)d_in[1];   // [768,256]
    const float* bias = (const float*)d_in[2];   // [768]
    float* Out        = (float*)d_out;           // [4,4096,256]

    _Float16* Qh = (_Float16*)d_ws;                       // 16384*256 f16 (8 MB)
    _Float16* Kh = Qh + (size_t)NTOK * DIM;               // 8 MB
    _Float16* Vt = Kh + (size_t)NTOK * DIM;               // 8 MB, [b][d][s]

    // 1024 M-tiles x 48 N-tiles, 4 waves/block of 128 threads
    qkv_proj_kernel<<<(1024 * 48) / 4, 128, 0, stream>>>(X, W, bias, Qh, Kh, Vt);
    // 1024 query tiles, 2 tiles/wave, 4 waves/block
    flash_attn_kernel<<<1024 / 8, 128, 0, stream>>>(Qh, Kh, Vt, Out);
}
